// AxialSpaMamba_60713657696377
// MI455X (gfx1250) — matrix operand order repaired and checked by reading, not compiled, not run on hardware
//
#include <hip/hip_runtime.h>
#include <math.h>

typedef __attribute__((ext_vector_type(16))) __bf16 v16bf;
typedef __attribute__((ext_vector_type(8)))  __bf16 v8bf;
typedef __attribute__((ext_vector_type(8)))  float  v8f;

#define SEQS 1024      // sequences per axis (b*h or b*w)
#define LSEQ 128       // sequence length
#define CIN  64        // model channels
#define DI   128       // inner dim (2*CIN)
#define DTR  4         // dt rank
#define NST  16        // D_STATE
#define NPROJ 36       // dtr + 2*16
#define NPAD  48       // padded to 3 N-tiles
#define MROWS (SEQS*LSEQ)   // 131072

__device__ __forceinline__ float dev_silu(float x) {
    return x / (1.0f + __expf(-x));
}
__device__ __forceinline__ float dev_softplus(float x) {
    return (x > 20.0f) ? x : log1pf(__expf(x));
}

// ---------------------------------------------------------------------------
// WMMA bf16 fragment loaders — unconditional b128 vector loads.
// A 16x32: lane<16 -> M=lane, slots 0..7=K(kk+half*8 ..), slots 8..15=K(+16)
// B 32x16: lane<16 -> N=lane, slots 0..7=K(kk+half*8 ..), slots 8..15=K(+16)
// ---------------------------------------------------------------------------
__device__ __forceinline__ v16bf load_a_frag(const __bf16* __restrict__ A,
                                             int lda, int m0, int kk) {
    int lane = threadIdx.x & 31;
    int half = lane >> 4;
    int m = m0 + (lane & 15);
    const __bf16* p = A + (size_t)m * lda + kk + half * 8;
    v8bf lo = *(const v8bf*)p;
    v8bf hi = *(const v8bf*)(p + 16);
    return __builtin_shufflevector(lo, hi, 0, 1, 2, 3, 4, 5, 6, 7,
                                   8, 9, 10, 11, 12, 13, 14, 15);
}

__device__ __forceinline__ v16bf load_b_frag(const __bf16* __restrict__ W,
                                             int ldw, int n0, int kk) {
    int lane = threadIdx.x & 31;
    int half = lane >> 4;
    int n = n0 + (lane & 15);
    const __bf16* p = W + (size_t)n * ldw + kk + half * 8;
    v8bf lo = *(const v8bf*)p;
    v8bf hi = *(const v8bf*)(p + 16);
    return __builtin_shufflevector(lo, hi, 0, 1, 2, 3, 4, 5, 6, 7,
                                   8, 9, 10, 11, 12, 13, 14, 15);
}

// ---------------------------------------------------------------------------
// Pack weights (Nrows x K f32, row-major) -> (Npad x K bf16), zero padded.
// ---------------------------------------------------------------------------
__global__ void k_pack_w(const float* __restrict__ W, __bf16* __restrict__ out,
                         int Nrows, int Npad, int K) {
    int i = blockIdx.x * blockDim.x + threadIdx.x;
    if (i >= Npad * K) return;
    int n = i / K;
    int k = i - n * K;
    out[i] = (__bf16)((n < Nrows) ? W[n * K + k] : 0.0f);
}

// ---------------------------------------------------------------------------
// Gather (b,c,h,w) -> sequence-major (s, t, c) bf16 for an axis.
// ---------------------------------------------------------------------------
__global__ void k_gather(const float* __restrict__ x, __bf16* __restrict__ xseq,
                         int axis) {
    size_t i = (size_t)blockIdx.x * blockDim.x + threadIdx.x;
    if (i >= (size_t)MROWS * CIN) return;
    int c = (int)(i & 63);
    size_t st = i >> 6;
    int t = (int)(st & 127);
    int s = (int)(st >> 7);
    int b = s >> 7;
    int q = s & 127;
    size_t src = (axis == 0)
        ? ((((size_t)b * CIN + c) * 128 + q) * 128 + t)
        : ((((size_t)b * CIN + c) * 128 + t) * 128 + q);
    xseq[i] = (__bf16)x[src];
}

// ---------------------------------------------------------------------------
// GEMM: C[M x Npad] = A[M x K] * W[Npad x K]^T.  bf16 in, f32 out.
// One wave computes NT consecutive 16x16 N-tiles of one M-tile (A reuse).
// K and NT are compile-time -> fully unrolled WMMA chain, no branches.
// ---------------------------------------------------------------------------
template <int K, int NT>
__global__ void k_gemm_bf(const __bf16* __restrict__ A, int lda,
                          const __bf16* __restrict__ W, int ldw,
                          float* __restrict__ C, int ldc, int Npad) {
    int wave = blockIdx.x * (blockDim.x >> 5) + (threadIdx.x >> 5);
    int ngrp = Npad / (16 * NT);
    int mt = wave / ngrp;
    int n0 = (wave - mt * ngrp) * (16 * NT);

    v8f acc[NT] = {};
#pragma unroll
    for (int kk = 0; kk < K; kk += 32) {
        v16bf a = load_a_frag(A, lda, mt * 16, kk);
#pragma unroll
        for (int j = 0; j < NT; ++j) {
            v16bf b = load_b_frag(W, ldw, n0 + j * 16, kk);
            acc[j] = __builtin_amdgcn_wmma_f32_16x16x32_bf16(
                false, a, false, b, (short)0, acc[j], false, false);
        }
    }

    int lane = threadIdx.x & 31;
    int half = lane >> 4;
#pragma unroll
    for (int j = 0; j < NT; ++j) {
        int col = n0 + j * 16 + (lane & 15);
#pragma unroll
        for (int r = 0; r < 8; ++r) {
            int row = mt * 16 + r + half * 8;
            C[(size_t)row * ldc + col] = acc[j][r];
        }
    }
}

// ---------------------------------------------------------------------------
// Out-projection GEMM (K=128, N=64) with axis-dependent scatter into x_rec,
// scaled by 0.5 (axis average). axis==1 accumulates on top of axis==0.
// ---------------------------------------------------------------------------
__global__ void k_gemm_out(const __bf16* __restrict__ Y,
                           const __bf16* __restrict__ Wout,
                           float* __restrict__ xrec, int axis) {
    int mt = blockIdx.x * (blockDim.x >> 5) + (threadIdx.x >> 5);

    v8f acc[4] = {};
#pragma unroll
    for (int kk = 0; kk < DI; kk += 32) {
        v16bf a = load_a_frag(Y, DI, mt * 16, kk);
#pragma unroll
        for (int j = 0; j < 4; ++j) {
            v16bf b = load_b_frag(Wout, DI, j * 16, kk);
            acc[j] = __builtin_amdgcn_wmma_f32_16x16x32_bf16(
                false, a, false, b, (short)0, acc[j], false, false);
        }
    }

    int lane = threadIdx.x & 31;
    int half = lane >> 4;
#pragma unroll
    for (int j = 0; j < 4; ++j) {
        int c = j * 16 + (lane & 15);
#pragma unroll
        for (int r = 0; r < 8; ++r) {
            int row = mt * 16 + r + half * 8;
            int s = row >> 7;
            int t = row & 127;
            int b = s >> 7;
            int q = s & 127;
            size_t idx = (axis == 0)
                ? ((((size_t)b * CIN + c) * 128 + q) * 128 + t)
                : ((((size_t)b * CIN + c) * 128 + t) * 128 + q);
            float v = 0.5f * acc[j][r];
            if (axis == 0) xrec[idx] = v;
            else           xrec[idx] += v;
        }
    }
}

// ---------------------------------------------------------------------------
// Depthwise causal conv (k=4) fwd + anti-causal (reverse pass) + SiLU.
// Writes f32 (scan input) and bf16 (next GEMM's A operand).
// ---------------------------------------------------------------------------
__global__ void k_conv(const float* __restrict__ xz,
                       const float* __restrict__ conv_w,
                       const float* __restrict__ conv_b,
                       float* __restrict__ xcf, float* __restrict__ xcb,
                       __bf16* __restrict__ xcf_bf, __bf16* __restrict__ xcb_bf) {
    size_t i = (size_t)blockIdx.x * blockDim.x + threadIdx.x;
    if (i >= (size_t)MROWS * DI) return;
    int d = (int)(i & 127);
    size_t st = i >> 7;
    int t = (int)(st & 127);
    size_t s = st >> 7;

    const float* base = xz + (s * 128) * 256 + d;  // (s, tt, d) at base + tt*256
    float w0 = conv_w[d * 4 + 0], w1 = conv_w[d * 4 + 1];
    float w2 = conv_w[d * 4 + 2], w3 = conv_w[d * 4 + 3];
    float bb = conv_b[d];

    auto X = [&](int tt) -> float {
        return (tt >= 0 && tt < LSEQ) ? base[(size_t)tt * 256] : 0.0f;
    };
    float f = bb + w0 * X(t - 3) + w1 * X(t - 2) + w2 * X(t - 1) + w3 * X(t);
    float r = bb + w0 * X(t + 3) + w1 * X(t + 2) + w2 * X(t + 1) + w3 * X(t);
    f = dev_silu(f);
    r = dev_silu(r);
    xcf[i] = f;
    xcb[i] = r;
    xcf_bf[i] = (__bf16)f;
    xcb_bf[i] = (__bf16)r;
}

// ---------------------------------------------------------------------------
// Selective scan: one block = one sequence, 128 threads = channels.
// dt/B/C staged in LDS once (no barriers in the time loop).
// Writes gated 0.5 * y * silu(z) as bf16; dir=1 runs reversed and accumulates.
// ---------------------------------------------------------------------------
__global__ void k_scan(const float* __restrict__ xconv,
                       const float* __restrict__ xdbl,
                       const float* __restrict__ xz,
                       const float* __restrict__ dt_w,
                       const float* __restrict__ dt_b,
                       const float* __restrict__ A_log,
                       const float* __restrict__ Dp,
                       __bf16* __restrict__ ycomb, int dir) {
    __shared__ float sBC[LSEQ * NPROJ];  // 18 KB
    int s = blockIdx.x;
    int d = threadIdx.x;

    for (int i = d; i < LSEQ * NPROJ; i += DI) {
        int t = i / NPROJ;
        int cc = i - t * NPROJ;
        sBC[i] = xdbl[((size_t)(s * LSEQ + t)) * NPAD + cc];
    }
    __syncthreads();

    float Ad[NST];
#pragma unroll
    for (int n = 0; n < NST; ++n) Ad[n] = -__expf(A_log[d * NST + n]);
    float w0 = dt_w[d * DTR + 0], w1 = dt_w[d * DTR + 1];
    float w2 = dt_w[d * DTR + 2], w3 = dt_w[d * DTR + 3];
    float db = dt_b[d];
    float Dd = Dp[d];

    float h[NST];
#pragma unroll
    for (int n = 0; n < NST; ++n) h[n] = 0.0f;

    int t0 = dir ? (LSEQ - 1) : 0;
    int step = dir ? -1 : 1;
    for (int it = 0; it < LSEQ; ++it) {
        int t = t0 + step * it;
        const float* row = &sBC[t * NPROJ];
        float delta = dev_softplus(row[0] * w0 + row[1] * w1 +
                                   row[2] * w2 + row[3] * w3 + db);
        float u = xconv[((size_t)s * LSEQ + t) * DI + d];
        float du = delta * u;
        float y = 0.0f;
#pragma unroll
        for (int n = 0; n < NST; ++n) {
            float dA = __expf(delta * Ad[n]);
            h[n] = dA * h[n] + du * row[DTR + n];
            y += h[n] * row[DTR + NST + n];
        }
        y += u * Dd;
        float z = xz[((size_t)(s * LSEQ + t)) * 256 + DI + d];
        float v = 0.5f * y * dev_silu(z);
        size_t oi = ((size_t)s * LSEQ + t) * DI + d;
        if (dir) ycomb[oi] = (__bf16)((float)ycomb[oi] + v);
        else     ycomb[oi] = (__bf16)v;
    }
}

// ---------------------------------------------------------------------------
// GroupNorm stats: one block per (b, g), g = 4 groups of 16 channels.
// ---------------------------------------------------------------------------
__global__ void k_gn_stats(const float* __restrict__ xrec,
                           float* __restrict__ stats) {
    __shared__ float ssum[256], ssq[256];
    int bg = blockIdx.x;
    int b = bg >> 2;
    int g = bg & 3;
    int c0 = g * 16;
    const int N = 16 * 128 * 128;  // 262144
    float acc = 0.0f, acc2 = 0.0f;
    for (int i = threadIdx.x; i < N; i += 256) {
        int c = c0 + (i >> 14);
        int off = i & 16383;
        float v = xrec[(((size_t)b * CIN + c) << 14) + off];
        acc += v;
        acc2 += v * v;
    }
    ssum[threadIdx.x] = acc;
    ssq[threadIdx.x] = acc2;
    __syncthreads();
    for (int o = 128; o > 0; o >>= 1) {
        if (threadIdx.x < o) {
            ssum[threadIdx.x] += ssum[threadIdx.x + o];
            ssq[threadIdx.x] += ssq[threadIdx.x + o];
        }
        __syncthreads();
    }
    if (threadIdx.x == 0) {
        float inv = 1.0f / (float)N;
        float mu = ssum[0] * inv;
        float var = ssq[0] * inv - mu * mu;
        stats[bg * 2 + 0] = mu;
        stats[bg * 2 + 1] = rsqrtf(var + 1e-5f);
    }
}

__global__ void k_gn_apply(const float* __restrict__ xrec,
                           const float* __restrict__ x,
                           const float* __restrict__ gnw,
                           const float* __restrict__ gnb,
                           const float* __restrict__ stats,
                           float* __restrict__ out) {
    size_t i = (size_t)blockIdx.x * blockDim.x + threadIdx.x;
    if (i >= (size_t)8 * CIN * 128 * 128) return;
    int c = (int)((i >> 14) & 63);
    int b = (int)(i >> 20);
    int bg = b * 4 + (c >> 4);
    float mu = stats[bg * 2 + 0];
    float rstd = stats[bg * 2 + 1];
    float v = (xrec[i] - mu) * rstd * gnw[c] + gnb[c];
    out[i] = dev_silu(v) + x[i];
}

// ---------------------------------------------------------------------------
extern "C" void kernel_launch(void* const* d_in, const int* in_sizes, int n_in,
                              void* d_out, int out_size, void* d_ws, size_t ws_size,
                              hipStream_t stream) {
    const float* x = (const float*)d_in[0];
    const float* gn_w = (const float*)d_in[19];
    const float* gn_b = (const float*)d_in[20];

    // Workspace carve-up (all chunks 64B-aligned by construction).
    char* base = (char*)d_ws;
    size_t o = 0;
    auto carve = [&](size_t bytes) { char* p = base + o; o += bytes; return p; };
    float*  xz     = (float*)carve((size_t)MROWS * 256 * 4);   // 134 MB
    float*  xcf    = (float*)carve((size_t)MROWS * DI * 4);    // 67 MB
    float*  xcb    = (float*)carve((size_t)MROWS * DI * 4);    // 67 MB
    float*  xdf    = (float*)carve((size_t)MROWS * NPAD * 4);  // 25 MB
    float*  xdb    = (float*)carve((size_t)MROWS * NPAD * 4);  // 25 MB
    float*  xrec   = (float*)carve((size_t)MROWS * CIN * 4);   // 33.5 MB
    float*  stats  = (float*)carve(256);
    __bf16* xseq   = (__bf16*)carve((size_t)MROWS * CIN * 2);  // 16.8 MB
    __bf16* xcf_bf = (__bf16*)carve((size_t)MROWS * DI * 2);   // 33.5 MB
    __bf16* xcb_bf = (__bf16*)carve((size_t)MROWS * DI * 2);   // 33.5 MB
    __bf16* yc     = (__bf16*)carve((size_t)MROWS * DI * 2);   // 33.5 MB
    __bf16* w_in   = (__bf16*)carve((size_t)256 * CIN * 2);
    __bf16* w_xp   = (__bf16*)carve((size_t)NPAD * DI * 2);
    __bf16* w_out  = (__bf16*)carve((size_t)CIN * DI * 2);
    (void)ws_size; (void)in_sizes; (void)n_in; (void)out_size;

    for (int axis = 0; axis < 2; ++axis) {
        int p = 1 + axis * 9;
        const float* in_w   = (const float*)d_in[p + 0];  // (256, 64)
        const float* conv_w = (const float*)d_in[p + 1];  // (128, 1, 4)
        const float* conv_b = (const float*)d_in[p + 2];  // (128)
        const float* xp_w   = (const float*)d_in[p + 3];  // (36, 128)
        const float* dt_w   = (const float*)d_in[p + 4];  // (128, 4)
        const float* dt_b   = (const float*)d_in[p + 5];  // (128)
        const float* A_log  = (const float*)d_in[p + 6];  // (128, 16)
        const float* Dp     = (const float*)d_in[p + 7];  // (128)
        const float* out_w  = (const float*)d_in[p + 8];  // (64, 128)

        // 0) pack weights to bf16 (xp zero-padded 36 -> 48 rows)
        k_pack_w<<<(256 * CIN + 255) / 256, 256, 0, stream>>>(in_w, w_in, 256, 256, CIN);
        k_pack_w<<<(NPAD * DI + 255) / 256, 256, 0, stream>>>(xp_w, w_xp, NPROJ, NPAD, DI);
        k_pack_w<<<(CIN * DI + 255) / 256, 256, 0, stream>>>(out_w, w_out, CIN, CIN, DI);

        // 1) gather into sequence-major bf16
        k_gather<<<(MROWS * CIN) / 256, 256, 0, stream>>>(x, xseq, axis);

        // 2) in-projection: xz = xseq @ in_w^T  (M=131072, N=256, K=64, NT=4)
        {
            int waves = (MROWS / 16) * (256 / 64);
            k_gemm_bf<64, 4><<<waves / 8, 256, 0, stream>>>(xseq, CIN, w_in, CIN,
                                                            xz, 256, 256);
        }

        // 3) depthwise conv (fwd + reverse-aligned) + SiLU
        k_conv<<<(MROWS * DI) / 256, 256, 0, stream>>>(xz, conv_w, conv_b,
                                                       xcf, xcb, xcf_bf, xcb_bf);

        // 4) x-projection both directions (M x 48, K=128, NT=3)
        {
            int waves = (MROWS / 16);
            k_gemm_bf<128, 3><<<waves / 8, 256, 0, stream>>>(xcf_bf, DI, w_xp, DI,
                                                             xdf, NPAD, NPAD);
            k_gemm_bf<128, 3><<<waves / 8, 256, 0, stream>>>(xcb_bf, DI, w_xp, DI,
                                                             xdb, NPAD, NPAD);
        }

        // 5) selective scans: fwd writes, bwd accumulates (gated, 0.5x each)
        k_scan<<<SEQS, DI, 0, stream>>>(xcf, xdf, xz, dt_w, dt_b, A_log, Dp, yc, 0);
        k_scan<<<SEQS, DI, 0, stream>>>(xcb, xdb, xz, dt_w, dt_b, A_log, Dp, yc, 1);

        // 6) out-projection with scatter into x_rec (0.5x axis average)
        k_gemm_out<<<(MROWS / 16) / 8, 256, 0, stream>>>(yc, w_out, xrec, axis);
    }

    // GroupNorm (g=4) + SiLU + residual
    k_gn_stats<<<32, 256, 0, stream>>>(xrec, stats);
    k_gn_apply<<<(8 * CIN * 128 * 128) / 256, 256, 0, stream>>>(
        xrec, x, gn_w, gn_b, stats, (float*)d_out);
}
